// Network_39797166964863
// MI455X (gfx1250) — compile-verified, matrix operand order checked by reading, not run on hardware
//
#include <hip/hip_runtime.h>
#include <math.h>

// ---------------------------------------------------------------------------
// CDNA5 (gfx1250) fused implementation.
// Precision: bf16 inputs -> v_wmma_f32_16x16x32_bf16 (f32 accumulate).
// All GEMM-like ops (ROI-fused 1x1 conv, FC heads, radar convs via im2col
// gather, radar-ROI GEMM) run on 16x16 WMMA tiles, one wave per tile.
// ---------------------------------------------------------------------------

typedef __attribute__((ext_vector_type(16))) __bf16 v16bf;
typedef __attribute__((ext_vector_type(8)))  __bf16 v8bf;
typedef __attribute__((ext_vector_type(8)))  float  v8f;

__device__ __forceinline__ __bf16 f2bf(float f) {
  union { float f; unsigned u; } x; x.f = f;
  unsigned r = (x.u + 0x7FFFu + ((x.u >> 16) & 1u)) >> 16;   // RNE
  union { unsigned short s; __bf16 b; } y; y.s = (unsigned short)r;
  return y.b;
}
__device__ __forceinline__ float leakyf(float v)   { return v > 0.f ? v : 0.1f * v; }
__device__ __forceinline__ float sigmoidf_(float v){ return 1.f / (1.f + __expf(-v)); }
__device__ __forceinline__ int   clampi(int v,int lo,int hi){ return v<lo?lo:(v>hi?hi:v); }
// 16-bit A/B WMMA K mapping per lane-group g (0: lanes 0-15, 1: lanes 16-31):
// slots 0..7 -> K = g*8 + s ; slots 8..15 -> K = 16 + g*8 + (s-8)
__device__ __forceinline__ int kmap(int g, int s) { return s < 8 ? g * 8 + s : 16 + g * 8 + (s - 8); }

__device__ __forceinline__ v8f wmma_bf16(v16bf a, v16bf b, v8f c) {
  return __builtin_amdgcn_wmma_f32_16x16x32_bf16(false, a, false, b, (short)0, c, false, false);
}

// ------------------------------ repack kernels ------------------------------

// features f32 NCHW (16,256,52,52) -> bf16 NHWC [b][y][x][c]
__global__ void pack_feat(const float* __restrict__ f, __bf16* __restrict__ o) {
  int idx = blockIdx.x * 256 + threadIdx.x;
  if (idx >= 16 * 256 * 52 * 52) return;
  int x = idx % 52, y = (idx / 52) % 52, c = (idx / 2704) % 256, b = idx / (2704 * 256);
  o[(((b * 52 + y) * 52 + x) * 256) + c] = f2bf(f[idx]);
}
// maps f32 NCHW (16,3,32,32) -> bf16 NHWC [b][y][x][c], C=3
__global__ void pack_maps(const float* __restrict__ f, __bf16* __restrict__ o) {
  int idx = blockIdx.x * 256 + threadIdx.x;
  if (idx >= 16 * 3 * 1024) return;
  int x = idx & 31, y = (idx >> 5) & 31, c = (idx >> 10) % 3, b = idx / 3072;
  o[((b * 32 + y) * 32 + x) * 3 + c] = f2bf(f[idx]);
}

// BN fold: s = g*rsqrt(v+eps); t = (conv_bias - m)*s + beta
__global__ void bn_prep(const float* g, const float* beta, const float* m, const float* v,
                        const float* cb, float* s, float* t, int n) {
  int i = blockIdx.x * 256 + threadIdx.x;
  if (i >= n) return;
  float si = g ? g[i] * rsqrtf(v[i] + 1e-5f) : 1.f;
  s[i] = si;
  t[i] = g ? (cb[i] - m[i]) * si + beta[i] : cb[i];
}

__global__ void build_h2bias(const float* b1, const float* b2, float* t) {
  int i = threadIdx.x;
  if (i < 32) t[i] = i < 4 ? b1[i] : (i < 17 ? b2[i - 4] : 0.f);
}

// Generic B packer: out[ntile][Kchunks][32 lanes][16 slots] (bf16).
// Weight layout: W[o][c][tap] flattened as o*(Cin*taps) + c*taps + tap,
// with K ordering k = tap*Cin + c (tap-major). Scale s[o] folded if non-null.
__global__ void pack_B(const float* __restrict__ W, int Cin, int taps, int Kreal, int Kchunks,
                       int Nreal, int nOffset, const float* __restrict__ s,
                       __bf16* __restrict__ out) {
  int tid = blockIdx.x * 256 + threadIdx.x;
  int total = Nreal * Kreal;
  if (tid >= total) return;
  int o = tid / Kreal, k = tid % Kreal;
  int n = nOffset + o;
  int kc = k >> 5, kk = k & 31;
  int g_, sl;
  if (kk < 16) { g_ = kk >> 3; sl = kk & 7; }
  else         { int q = kk - 16; g_ = q >> 3; sl = 8 + (q & 7); }
  int lane = g_ * 16 + (n & 15);
  int ntile = n >> 4;
  int c = k % Cin, tap = k / Cin;
  float w = W[(size_t)o * Cin * taps + c * taps + tap];
  if (s) w *= s[o];
  out[((size_t)(ntile * Kchunks + kc) * 32 + lane) * 16 + sl] = f2bf(w);
}

// img_cnn weight packer: per (i,j) bin, 10 channel rows {c*49+ij}, K=256.
// out[ij][8 kc][32][16]
__global__ void pack_Bimg(const float* __restrict__ W, const float* __restrict__ sI,
                          __bf16* __restrict__ out) {
  int tid = blockIdx.x * 256 + threadIdx.x;
  if (tid >= 49 * 10 * 256) return;
  int k = tid & 255;
  int rest = tid >> 8;
  int c = rest % 10, ij = rest / 10;
  int ch = c * 49 + ij;
  int kc = k >> 5, kk = k & 31;
  int g_, sl;
  if (kk < 16) { g_ = kk >> 3; sl = kk & 7; }
  else         { int q = kk - 16; g_ = q >> 3; sl = 8 + (q & 7); }
  int lane = g_ * 16 + c;
  out[((size_t)(ij * 8 + kc) * 32 + lane) * 16 + sl] = f2bf(W[(size_t)ch * 256 + k] * sI[ch]);
}

// ------------------------------ GEMM kernels --------------------------------

// Fused ROI gather + 1x1 conv + BN + leaky.  One wave = 16 boxes x 16 cols
// (10 real channels) for one (i,j) bin.  K = 256 (8 chunks).
__global__ void roi_gemm(const __bf16* __restrict__ feat, const float* __restrict__ box,
                         const int* __restrict__ bbatch, const __bf16* __restrict__ Bimg,
                         const float* __restrict__ tI, __bf16* __restrict__ flat) {
  int ij = blockIdx.y * blockDim.y + threadIdx.y;
  if (ij >= 49) return;
  int mtile = blockIdx.x;
  int lane = threadIdx.x;
  int m = lane & 15, g = lane >> 4;
  int n = mtile * 16 + m;
  int b = bbatch[n];
  float x1 = box[4 * n], y1 = box[4 * n + 1], x2 = box[4 * n + 2], y2 = box[4 * n + 3];
  int i = ij / 7, j = ij % 7;
  float ty = (i + 0.5f) / 7.0f, tx = (j + 0.5f) / 7.0f;
  int yy = clampi((int)((y1 + ty * (y2 - y1)) * 52.0f), 0, 51);
  int xx = clampi((int)((x1 + tx * (x2 - x1)) * 52.0f), 0, 51);
  const __bf16* arow = feat + ((size_t)(b * 52 + yy) * 52 + xx) * 256;
  v8f acc;
#pragma unroll
  for (int q = 0; q < 8; ++q) acc[q] = 0.f;
  for (int kc = 0; kc < 8; ++kc) {
    v8bf a0 = *(const v8bf*)(arow + kc * 32 + g * 8);
    v8bf a1 = *(const v8bf*)(arow + kc * 32 + 16 + g * 8);
    v16bf a;
#pragma unroll
    for (int t = 0; t < 8; ++t) { a[t] = a0[t]; a[8 + t] = a1[t]; }
    const __bf16* bp = Bimg + ((size_t)(ij * 8 + kc) * 32 + lane) * 16;
    if (kc < 7) __builtin_prefetch((const void*)(bp + 32 * 16), 0, 3);  // WGP-scope
    v16bf bvv = *(const v16bf*)bp;
    acc = wmma_bf16(a, bvv, acc);
  }
  int ncol = lane & 15;
  if (ncol < 10) {
    float tv = tI[ncol * 49 + ij];
#pragma unroll
    for (int v = 0; v < 8; ++v) {
      int n2 = mtile * 16 + v + 8 * g;
      flat[(size_t)n2 * 512 + ncol * 49 + ij] = f2bf(leakyf(acc[v] + tv));
    }
  }
}

// Row-major A GEMM.  MODE 0: leaky -> bf16 row-major.  MODE 1: raw -> f32.
template <int MODE>
__global__ void rm_gemm(const __bf16* __restrict__ A, int Kpad, int Kchunks,
                        const __bf16* __restrict__ Bpk, const float* __restrict__ tvec,
                        int Ncols, int ntilesTotal,
                        __bf16* __restrict__ outbf, float* __restrict__ outf) {
  int ntile = blockIdx.y * blockDim.y + threadIdx.y;
  if (ntile >= ntilesTotal) return;
  int mtile = blockIdx.x, lane = threadIdx.x;
  int m = lane & 15, g = lane >> 4;
  const __bf16* arow = A + (size_t)(mtile * 16 + m) * Kpad;
  v8f acc;
#pragma unroll
  for (int q = 0; q < 8; ++q) acc[q] = 0.f;
  for (int kc = 0; kc < Kchunks; ++kc) {
    v8bf a0 = *(const v8bf*)(arow + kc * 32 + g * 8);
    v8bf a1 = *(const v8bf*)(arow + kc * 32 + 16 + g * 8);
    v16bf a;
#pragma unroll
    for (int t = 0; t < 8; ++t) { a[t] = a0[t]; a[8 + t] = a1[t]; }
    v16bf bvv = *(const v16bf*)(Bpk + ((size_t)(ntile * Kchunks + kc) * 32 + lane) * 16);
    acc = wmma_bf16(a, bvv, acc);
  }
  int ncol = lane & 15, col = ntile * 16 + ncol;
  float tv = tvec[col];
#pragma unroll
  for (int v = 0; v < 8; ++v) {
    int n2 = mtile * 16 + v + 8 * g;
    float xv = acc[v] + tv;
    if (MODE == 0) outbf[(size_t)n2 * Ncols + col] = f2bf(leakyf(xv));
    else           outf[(size_t)n2 * Ncols + col] = xv;
  }
}

// Generic radar conv as im2col-gather GEMM (scalar gather; used only for
// conv1 where Cin=3 so K-runs are not contiguous).
template <int MODE>
__global__ void conv_gemm(const __bf16* __restrict__ act, int Cin, int taps, int Kreal,
                          int Kchunks, const __bf16* __restrict__ Bpk,
                          const float* __restrict__ tvec, int Creal, int Cout,
                          int ntilesTotal, __bf16* __restrict__ out_bf,
                          float* __restrict__ out_f) {
  int ntile = blockIdx.y * blockDim.y + threadIdx.y;
  if (ntile >= ntilesTotal) return;
  int mtile = blockIdx.x, lane = threadIdx.x;
  int m = lane & 15, g = lane >> 4;
  int p = mtile * 16 + m;
  int b = p >> 10, y = (p >> 5) & 31, x = p & 31;
  v8f acc;
#pragma unroll
  for (int q = 0; q < 8; ++q) acc[q] = 0.f;
  for (int kc = 0; kc < Kchunks; ++kc) {
    v16bf a;
#pragma unroll
    for (int s = 0; s < 16; ++s) {
      int k = kc * 32 + kmap(g, s);
      __bf16 val = f2bf(0.f);
      if (k < Kreal) {
        int tap = (taps == 1) ? 0 : (k / Cin);
        int c   = (taps == 1) ? k : (k % Cin);
        int dy  = (taps == 1) ? 0 : (tap / 3 - 1);
        int dx  = (taps == 1) ? 0 : (tap % 3 - 1);
        int ys = y + dy, xs = x + dx;
        if (ys >= 0 && ys < 32 && xs >= 0 && xs < 32)
          val = act[((size_t)(b * 32 + ys) * 32 + xs) * Cin + c];
      }
      a[s] = val;
    }
    v16bf bvv = *(const v16bf*)(Bpk + ((size_t)(ntile * Kchunks + kc) * 32 + lane) * 16);
    acc = wmma_bf16(a, bvv, acc);
  }
  int ncol = lane & 15;
  if (MODE == 0) {
    int col = ntile * 16 + ncol;
    float tv = (col < Creal) ? tvec[col] : 0.f;
#pragma unroll
    for (int v = 0; v < 8; ++v) {
      int p2 = mtile * 16 + v + 8 * g;
      out_bf[(size_t)p2 * Cout + col] = f2bf(leakyf(acc[v] + tv));
    }
  } else {
    if (ncol < Creal) {
      float tv = tvec[ncol];
#pragma unroll
      for (int v = 0; v < 8; ++v) {
        int p2 = mtile * 16 + v + 8 * g;
        int b2 = p2 >> 10, y2 = (p2 >> 5) & 31, x2 = p2 & 31;
        out_f[(((size_t)b2 * Creal + ncol) << 10) + (y2 << 5) + x2] = sigmoidf_(acc[v] + tv);
      }
    }
  }
}

// Vectorized radar conv GEMM for CIN multiple of 32 (NHWC): each 32-wide
// K-chunk is one (tap, 32-channel half), so A loads are two b128s per lane.
// MODE 0: BN+leaky -> bf16 NHWC.  MODE 1: sigmoid -> f32 NCHW (Creal chans).
template <int MODE, int CIN, int TAPS>
__global__ void conv_gemm_v(const __bf16* __restrict__ act, const __bf16* __restrict__ Bpk,
                            const float* __restrict__ tvec, int Creal, int Cout,
                            int ntilesTotal, __bf16* __restrict__ out_bf,
                            float* __restrict__ out_f) {
  constexpr int HALVES = CIN / 32;
  constexpr int KCH = HALVES * TAPS;
  int ntile = blockIdx.y * blockDim.y + threadIdx.y;
  if (ntile >= ntilesTotal) return;
  int mtile = blockIdx.x, lane = threadIdx.x;
  int m = lane & 15, g = lane >> 4;
  int p = mtile * 16 + m;
  int b = p >> 10, y = (p >> 5) & 31, x = p & 31;
  v8f acc;
#pragma unroll
  for (int q = 0; q < 8; ++q) acc[q] = 0.f;
#pragma unroll
  for (int kc = 0; kc < KCH; ++kc) {
    const int tap = kc / HALVES;
    const int cbase = (kc % HALVES) * 32;
    const int dy = (TAPS == 1) ? 0 : (tap / 3 - 1);
    const int dx = (TAPS == 1) ? 0 : (tap % 3 - 1);
    int ys = y + dy, xs = x + dx;
    bool ok = (TAPS == 1) || (ys >= 0 && ys < 32 && xs >= 0 && xs < 32);
    v16bf a;
    if (ok) {
      const __bf16* ap = act + ((size_t)(b * 32 + ys) * 32 + xs) * CIN + cbase;
      v8bf a0 = *(const v8bf*)(ap + g * 8);
      v8bf a1 = *(const v8bf*)(ap + 16 + g * 8);
#pragma unroll
      for (int t = 0; t < 8; ++t) { a[t] = a0[t]; a[8 + t] = a1[t]; }
    } else {
      __bf16 z = f2bf(0.f);
#pragma unroll
      for (int t = 0; t < 16; ++t) a[t] = z;
    }
    const __bf16* bp = Bpk + ((size_t)(ntile * KCH + kc) * 32 + lane) * 16;
    if (kc + 1 < KCH) __builtin_prefetch((const void*)(bp + 32 * 16), 0, 3);
    v16bf bvv = *(const v16bf*)bp;
    acc = wmma_bf16(a, bvv, acc);
  }
  int ncol = lane & 15;
  if (MODE == 0) {
    int col = ntile * 16 + ncol;
    float tv = (col < Creal) ? tvec[col] : 0.f;
#pragma unroll
    for (int v = 0; v < 8; ++v) {
      int p2 = mtile * 16 + v + 8 * g;
      out_bf[(size_t)p2 * Cout + col] = f2bf(leakyf(acc[v] + tv));
    }
  } else {
    if (ncol < Creal) {
      float tv = tvec[ncol];
#pragma unroll
      for (int v = 0; v < 8; ++v) {
        int p2 = mtile * 16 + v + 8 * g;
        int b2 = p2 >> 10, y2 = (p2 >> 5) & 31, x2 = p2 & 31;
        out_f[(((size_t)b2 * Creal + ncol) << 10) + (y2 << 5) + x2] = sigmoidf_(acc[v] + tv);
      }
    }
  }
}

// Radar-ROI fused gather + (10,10,7,7) VALID conv as GEMM: M=2048, K=490->512, N=16.
__global__ void rroi_gemm(const float* __restrict__ rs, const float* __restrict__ box,
                          const int* __restrict__ bbatch, const __bf16* __restrict__ Brw1,
                          const float* __restrict__ tR, float* __restrict__ rout) {
  int mtile = blockIdx.x, lane = threadIdx.x;
  int m = lane & 15, g = lane >> 4;
  int n = mtile * 16 + m;
  int b = bbatch[n];
  float x1 = box[4 * n], y1 = box[4 * n + 1], x2 = box[4 * n + 2], y2 = box[4 * n + 3];
  v8f acc;
#pragma unroll
  for (int q = 0; q < 8; ++q) acc[q] = 0.f;
  for (int kc = 0; kc < 16; ++kc) {
    v16bf a;
#pragma unroll
    for (int s = 0; s < 16; ++s) {
      int k = kc * 32 + kmap(g, s);
      float val = 0.f;
      if (k < 490) {
        int c = k / 49, r = k % 49, i = r / 7, j = r % 7;
        float ty = (i + 0.5f) / 7.0f, tx = (j + 0.5f) / 7.0f;
        int yy = clampi((int)((y1 + ty * (y2 - y1)) * 32.0f), 0, 31);
        int xx = clampi((int)((x1 + tx * (x2 - x1)) * 32.0f), 0, 31);
        val = rs[(((size_t)b * 10 + c) << 10) + (yy << 5) + xx];
      }
      a[s] = f2bf(val);
    }
    v16bf bvv = *(const v16bf*)(Brw1 + ((size_t)(kc * 32 + lane)) * 16);
    acc = wmma_bf16(a, bvv, acc);
  }
  int ncol = lane & 15;
  if (ncol < 10) {
    float tv = tR[ncol];
#pragma unroll
    for (int v = 0; v < 8; ++v) {
      int n2 = mtile * 16 + v + 8 * g;
      rout[(size_t)n2 * 16 + ncol] = leakyf(acc[v] + tv);
    }
  }
}

// Per-box tail: rw2 1x1 + sigmoid, confidence, ensemble MLP, softmax, concat.
__global__ void final_ens(const float* __restrict__ h2, const float* __restrict__ rout,
                          const float* __restrict__ rw2, const float* __restrict__ rb2,
                          const float* __restrict__ ew1, const float* __restrict__ eb1,
                          const float* __restrict__ ew2, const float* __restrict__ eb2,
                          const float* __restrict__ yolo, float* __restrict__ out) {
  int n = blockIdx.x * blockDim.x + threadIdx.x;
  if (n >= 2048) return;
  const float* h = h2 + (size_t)n * 32;
  float class0 = sigmoidf_(h[4]);
  float class1 = sigmoidf_(h[5]);
  float rc = rb2[0];
#pragma unroll
  for (int c = 0; c < 10; ++c) rc += rout[(size_t)n * 16 + c] * rw2[c];
  rc = sigmoidf_(rc);
  float conf = sigmoidf_(rc + class0);
  float refv[2] = {conf, class1};
  float yv[2] = {yolo[2 * n], yolo[2 * n + 1]};
  float l0 = eb2[0], l1 = eb2[1];
#pragma unroll
  for (int k = 0; k < 2; ++k) {
    for (int o = 0; o < 32; ++o) {
      float e = leakyf(refv[k] * ew1[2 * o] + yv[k] * ew1[2 * o + 1] + eb1[o]);
      int idx = k * 32 + o;
      l0 += e * ew2[idx];
      l1 += e * ew2[64 + idx];
    }
  }
  float mx = fmaxf(l0, l1);
  float e0 = __expf(l0 - mx), e1 = __expf(l1 - mx);
  float inv = 1.f / (e0 + e1);
  out[n * 6 + 0] = h[0]; out[n * 6 + 1] = h[1];
  out[n * 6 + 2] = h[2]; out[n * 6 + 3] = h[3];
  out[n * 6 + 4] = e0 * inv; out[n * 6 + 5] = e1 * inv;
}

// ------------------------------ host side -----------------------------------

// Canonical param slots
enum {
  IMG_W, IMG_B, IMG_G, IMG_BB, IMG_M, IMG_V,
  R_W1, R_B1, BN1G, BN1B, BN1M, BN1V,
  R_W2, R_B2, BN2G, BN2B, BN2M, BN2V,
  R_W3, R_B3, BN3G, BN3B, BN3M, BN3V,
  R_W4, R_B4,
  REF_W0, REF_B0, REF_W1, REF_B1, REF_W2, REF_B2,
  RRW1, RRB1, RBNG, RBNB, RBNM, RBNV,
  RRW2, RRB2,
  ENS_W1, ENS_B1, ENS_W2, ENS_B2, NPAR
};
static const int IDX_INS[NPAR] = {5,6,7,8,9,10, 11,12,13,14,15,16, 17,18,19,20,21,22,
                                  23,24,25,26,27,28, 29,30, 31,32,33,34,35,36,
                                  37,38,39,40,41,42, 43,44, 45,46,47,48};
static const int IDX_SRT[NPAR] = {14,9,11,10,12,13, 31,15,20,19,21,22, 32,16,24,23,25,26,
                                  33,17,28,27,29,30, 34,18, 46,35,47,36,48,37,
                                  44,38,41,40,42,43, 45,39, 7,5,8,6};

extern "C" void kernel_launch(void* const* d_in, const int* in_sizes, int n_in,
                              void* d_out, int out_size, void* d_ws, size_t ws_size,
                              hipStream_t stream) {
  const float* features = (const float*)d_in[0];
  const float* maps     = (const float*)d_in[1];
  const float* box      = (const float*)d_in[2];
  const int*   bbatch   = (const int*)d_in[3];
  const float* yolo     = (const float*)d_in[4];

  // params flattening order: dict-insertion vs jax tree (sorted keys)
  const int* tab = (n_in > 5 && in_sizes[5] == 125440) ? IDX_INS : IDX_SRT;
  const float* P[NPAR];
  for (int i = 0; i < NPAR; ++i) P[i] = (const float*)d_in[tab[i]];

  // ---- workspace carve (256B aligned) ----
  size_t off = 0;
  char* base = (char*)d_ws;
  auto carve = [&](size_t bytes) -> char* {
    char* p = base + off;
    off += (bytes + 255) & ~(size_t)255;
    return p;
  };
  __bf16* FEAT  = (__bf16*)carve((size_t)16 * 52 * 52 * 256 * 2);
  char* bz0 = base + off;                                   // start of packed-B zone
  __bf16* BIMG  = (__bf16*)carve((size_t)49 * 8 * 32 * 16 * 2);
  __bf16* BW0   = (__bf16*)carve((size_t)16 * 16 * 32 * 16 * 2);
  __bf16* BH2   = (__bf16*)carve((size_t)2 * 8 * 32 * 16 * 2);
  __bf16* BC1   = (__bf16*)carve((size_t)2 * 1 * 32 * 16 * 2);
  __bf16* BC2   = (__bf16*)carve((size_t)4 * 9 * 32 * 16 * 2);
  __bf16* BC3   = (__bf16*)carve((size_t)8 * 18 * 32 * 16 * 2);
  __bf16* BC4   = (__bf16*)carve((size_t)1 * 4 * 32 * 16 * 2);
  __bf16* BRW1  = (__bf16*)carve((size_t)1 * 16 * 32 * 16 * 2);
  size_t bzBytes = (size_t)((base + off) - bz0);
  __bf16* FLAT   = (__bf16*)carve((size_t)2048 * 512 * 2);
  __bf16* IMGTMP = (__bf16*)carve((size_t)2048 * 256 * 2);
  float*  H2OUT  = (float*)carve((size_t)2048 * 32 * 4);
  __bf16* MAPSB  = (__bf16*)carve((size_t)16 * 1024 * 3 * 2);
  __bf16* R1     = (__bf16*)carve((size_t)16 * 1024 * 32 * 2);
  __bf16* R2     = (__bf16*)carve((size_t)16 * 1024 * 64 * 2);
  __bf16* R3     = (__bf16*)carve((size_t)16 * 1024 * 128 * 2);
  float*  RS     = (float*)carve((size_t)16 * 10 * 1024 * 4);
  float*  ROUT   = (float*)carve((size_t)2048 * 16 * 4);
  float*  SI  = (float*)carve(490 * 4);
  float*  TI  = (float*)carve(490 * 4);
  float*  S1  = (float*)carve(32 * 4);
  float*  T1  = (float*)carve(32 * 4);
  float*  S2  = (float*)carve(64 * 4);
  float*  T2  = (float*)carve(64 * 4);
  float*  S3  = (float*)carve(128 * 4);
  float*  T3  = (float*)carve(128 * 4);
  float*  SR  = (float*)carve(16 * 4);
  float*  TR  = (float*)carve(16 * 4);
  float*  TH2 = (float*)carve(32 * 4);
  (void)ws_size; (void)out_size;

  // ---- stage 0: repack activations, zero packed-B zone, fold BN ----
  pack_feat<<<dim3((16 * 256 * 52 * 52 + 255) / 256), dim3(256), 0, stream>>>(features, FEAT);
  pack_maps<<<dim3((16 * 3 * 1024 + 255) / 256), dim3(256), 0, stream>>>(maps, MAPSB);
  hipMemsetAsync(bz0, 0, bzBytes, stream);

  bn_prep<<<dim3(2), dim3(256), 0, stream>>>(P[IMG_G], P[IMG_BB], P[IMG_M], P[IMG_V], P[IMG_B], SI, TI, 490);
  bn_prep<<<dim3(1), dim3(256), 0, stream>>>(P[BN1G], P[BN1B], P[BN1M], P[BN1V], P[R_B1], S1, T1, 32);
  bn_prep<<<dim3(1), dim3(256), 0, stream>>>(P[BN2G], P[BN2B], P[BN2M], P[BN2V], P[R_B2], S2, T2, 64);
  bn_prep<<<dim3(1), dim3(256), 0, stream>>>(P[BN3G], P[BN3B], P[BN3M], P[BN3V], P[R_B3], S3, T3, 128);
  bn_prep<<<dim3(1), dim3(256), 0, stream>>>(P[RBNG], P[RBNB], P[RBNM], P[RBNV], P[RRB1], SR, TR, 10);
  build_h2bias<<<dim3(1), dim3(32), 0, stream>>>(P[REF_B1], P[REF_B2], TH2);

  // ---- stage 1: weight packing (BN scale folded) ----
  pack_Bimg<<<dim3((49 * 10 * 256 + 255) / 256), dim3(256), 0, stream>>>(P[IMG_W], SI, BIMG);
  pack_B<<<dim3((256 * 490 + 255) / 256), dim3(256), 0, stream>>>(P[REF_W0], 490, 1, 490, 16, 256, 0, nullptr, BW0);
  pack_B<<<dim3((4 * 256 + 255) / 256), dim3(256), 0, stream>>>(P[REF_W1], 256, 1, 256, 8, 4, 0, nullptr, BH2);
  pack_B<<<dim3((13 * 256 + 255) / 256), dim3(256), 0, stream>>>(P[REF_W2], 256, 1, 256, 8, 13, 4, nullptr, BH2);
  pack_B<<<dim3((32 * 27 + 255) / 256), dim3(256), 0, stream>>>(P[R_W1], 3, 9, 27, 1, 32, 0, S1, BC1);
  pack_B<<<dim3((64 * 288 + 255) / 256), dim3(256), 0, stream>>>(P[R_W2], 32, 9, 288, 9, 64, 0, S2, BC2);
  pack_B<<<dim3((128 * 576 + 255) / 256), dim3(256), 0, stream>>>(P[R_W3], 64, 9, 576, 18, 128, 0, S3, BC3);
  pack_B<<<dim3((10 * 128 + 255) / 256), dim3(256), 0, stream>>>(P[R_W4], 128, 1, 128, 4, 10, 0, nullptr, BC4);
  pack_B<<<dim3((10 * 490 + 255) / 256), dim3(256), 0, stream>>>(P[RRW1], 490, 1, 490, 16, 10, 0, SR, BRW1);

  // ---- stage 2: radar CNN (WMMA; conv2-4 use vectorized b128 im2col) ----
  conv_gemm<0><<<dim3(1024, 1), dim3(32, 2), 0, stream>>>(MAPSB, 3, 9, 27, 1, BC1, T1, 32, 32, 2, R1, nullptr);
  conv_gemm_v<0, 32, 9><<<dim3(1024, 1), dim3(32, 4), 0, stream>>>(R1, BC2, T2, 64, 64, 4, R2, nullptr);
  conv_gemm_v<0, 64, 9><<<dim3(1024, 2), dim3(32, 4), 0, stream>>>(R2, BC3, T3, 128, 128, 8, R3, nullptr);
  conv_gemm_v<1, 128, 1><<<dim3(1024, 1), dim3(32, 1), 0, stream>>>(R3, BC4, P[R_B4], 10, 16, 1, nullptr, RS);

  // ---- stage 3: fused ROI gather + 1x1 conv GEMM (skips dense 85MB score_map) ----
  roi_gemm<<<dim3(128, 13), dim3(32, 4), 0, stream>>>(FEAT, box, bbatch, BIMG, TI, FLAT);

  // ---- stage 4: FC heads ----
  rm_gemm<0><<<dim3(128, 4), dim3(32, 4), 0, stream>>>(FLAT, 512, 16, BW0, P[REF_B0], 256, 16, IMGTMP, nullptr);
  rm_gemm<1><<<dim3(128, 1), dim3(32, 2), 0, stream>>>(IMGTMP, 256, 8, BH2, TH2, 32, 2, nullptr, H2OUT);

  // ---- stage 5: radar-ROI GEMM ----
  rroi_gemm<<<dim3(128), dim3(32), 0, stream>>>(RS, box, bbatch, BRW1, TR, ROUT);

  // ---- stage 6: per-box tail ----
  final_ens<<<dim3(8), dim3(256), 0, stream>>>(H2OUT, ROUT, P[RRW2], P[RRB2],
                                               P[ENS_W1], P[ENS_B1], P[ENS_W2], P[ENS_B2],
                                               yolo, (float*)d_out);
}
